// CausalSelfAttention_82540681495059
// MI455X (gfx1250) — matrix operand order repaired and checked by reading, not compiled, not run on hardware
//
// CDNA5 (gfx1250 / MI455X) causal self-attention, bf16 WMMA + async LDS copies:
//   prep:  x -> bf16, W_attn/W_proj -> bf16 transposed [N,K]
//   qkv  = x @ W_attn + b_attn        (bf16 WMMA, f32 accum, double-buffered LDS)
//   flash attention per head          (bf16 WMMA, online softmax in f32)
//   out  = y @ W_proj + b_proj        (bf16 WMMA, f32 accum, f32 output)
#include <hip/hip_runtime.h>
#include <cstdint>

#define BB  4
#define TT  2048
#define CC  768
#define NHH 12
#define DD  64

#define AS1 __attribute__((address_space(1)))
#define AS3 __attribute__((address_space(3)))

#if __has_builtin(__builtin_amdgcn_global_load_async_to_lds_b128)
#define HAVE_ASYNC 1
#else
#define HAVE_ASYNC 0
#endif

typedef __attribute__((ext_vector_type(16))) __bf16 v16bf;
typedef __attribute__((ext_vector_type(8)))  float  v8f;
typedef int v4i __attribute__((vector_size(16)));   // matches builtin param type

union Frag {
  uint4 u4[2];
  v16bf v;
};

__device__ __forceinline__ unsigned short f2bf(float f) {
  unsigned int u = __builtin_bit_cast(unsigned int, f);
  u += 0x7FFFu + ((u >> 16) & 1u);        // round-to-nearest-even
  return (unsigned short)(u >> 16);
}

__device__ __forceinline__ v8f wmma_bf16(const Frag& a, const Frag& b, v8f c) {
  return __builtin_amdgcn_wmma_f32_16x16x32_bf16(
      false, a.v, false, b.v, (short)0, c, false, false);
}

#if HAVE_ASYNC
__device__ __forceinline__ void copy_b128_lds(const unsigned short* g, unsigned short* l) {
  __builtin_amdgcn_global_load_async_to_lds_b128(
      (AS1 v4i*)g, (AS3 v4i*)l, 0, 0);
}
__device__ __forceinline__ void async_join() {
#if __has_builtin(__builtin_amdgcn_s_wait_asynccnt)
  __builtin_amdgcn_s_wait_asynccnt(0);
#else
  asm volatile("s_wait_asynccnt 0x0" ::: "memory");
#endif
}
#else
__device__ __forceinline__ void copy_b128_lds(const unsigned short* g, unsigned short* l) {
  *(uint4*)l = *(const uint4*)g;
}
__device__ __forceinline__ void async_join() {}
#endif

// ---------------------------------------------------------------------------
// Prep: f32 -> bf16 convert (vectorized x4)
// ---------------------------------------------------------------------------
__global__ __launch_bounds__(256)
void convert_f32_bf16(const float* __restrict__ src, unsigned short* __restrict__ dst,
                      int n4) {
  int i = blockIdx.x * 256 + threadIdx.x;
  if (i < n4) {
    float4 f = ((const float4*)src)[i];
    uint2 d;
    d.x = (unsigned)f2bf(f.x) | ((unsigned)f2bf(f.y) << 16);
    d.y = (unsigned)f2bf(f.z) | ((unsigned)f2bf(f.w) << 16);
    ((uint2*)dst)[i] = d;
  }
}

// ---------------------------------------------------------------------------
// Prep: W[K,N] f32 -> Wt[N,K] bf16, LDS-tiled 64x64 (coalesced both sides)
// ---------------------------------------------------------------------------
__global__ __launch_bounds__(256)
void transpose_f32_bf16(const float* __restrict__ W, unsigned short* __restrict__ Wt,
                        int K, int N) {
  __shared__ unsigned short t[64][65];
  const int n0 = blockIdx.x * 64, k0 = blockIdx.y * 64;
#pragma unroll
  for (int it = 0; it < 16; ++it) {
    int idx = threadIdx.x + it * 256;
    int r = idx >> 6, c = idx & 63;
    t[c][r] = f2bf(W[(size_t)(k0 + r) * N + n0 + c]);
  }
  __syncthreads();
#pragma unroll
  for (int it = 0; it < 16; ++it) {
    int idx = threadIdx.x + it * 256;
    int r = idx >> 6, c = idx & 63;
    Wt[(size_t)(n0 + r) * K + k0 + c] = t[r][c];
  }
}

// ---------------------------------------------------------------------------
// GEMM:  O[M,N] = A[M,K] * Bt[N,K]^T + bias[N]     (A, Bt bf16)
// Block tile 128x256, K-step 32, 8 waves, wave tile 64x64 (16 WMMAs/step).
// Double-buffered LDS; async global->LDS copies when available.
// ---------------------------------------------------------------------------
template <bool OUT_F32>
__global__ __launch_bounds__(256)
void gemm_tn_bias(const unsigned short* __restrict__ A,
                  const unsigned short* __restrict__ Bt,
                  const float* __restrict__ bias, void* __restrict__ Ov,
                  int M, int N, int K) {
  __shared__ unsigned short aT[2][128 * 32];   // [row][k]  8 KB each
  __shared__ unsigned short bT[2][256 * 32];   // [col][k] 16 KB each

  const int tid  = threadIdx.x;
  const int lane = tid & 31;
  const int wave = tid >> 5;
  const int wm   = wave >> 2;                  // 0..1 (64 rows)
  const int wn   = wave & 3;                   // 0..3 (64 cols)
  const int m0   = blockIdx.y * 128;
  const int n0   = blockIdx.x * 256;

  const int l15  = lane & 15;
  const int dsel = (lane < 16) ? 0 : 8;
  const int rsel = (lane < 16) ? 0 : 16;
  const int hsel = (lane >> 4) << 3;

  v8f acc[4][4];
#pragma unroll
  for (int i = 0; i < 4; ++i)
#pragma unroll
    for (int j = 0; j < 4; ++j) acc[i][j] = {};

  const int nk = K / 32;

  // ---- stage tile 0 ----
#pragma unroll
  for (int it = 0; it < 2; ++it) {            // A: 512 x b128 chunks
    int c = tid + it * 256;
    int row = c >> 2, off = (c & 3) * 8;
    copy_b128_lds(A + (size_t)(m0 + row) * K + off, &aT[0][row * 32 + off]);
  }
#pragma unroll
  for (int it = 0; it < 4; ++it) {            // Bt: 1024 x b128 chunks
    int c = tid + it * 256;
    int row = c >> 2, off = (c & 3) * 8;
    copy_b128_lds(Bt + (size_t)(n0 + row) * K + off, &bT[0][row * 32 + off]);
  }
  async_join();
  __syncthreads();

  for (int t = 0; t < nk; ++t) {
    const int ib = t & 1;
    const int kb1 = (t + 1) * 32;

#if HAVE_ASYNC
    if (t + 1 < nk) {                         // async prefetch of tile t+1
#pragma unroll
      for (int it = 0; it < 2; ++it) {
        int c = tid + it * 256;
        int row = c >> 2, off = (c & 3) * 8;
        copy_b128_lds(A + (size_t)(m0 + row) * K + kb1 + off,
                      &aT[ib ^ 1][row * 32 + off]);
      }
#pragma unroll
      for (int it = 0; it < 4; ++it) {
        int c = tid + it * 256;
        int row = c >> 2, off = (c & 3) * 8;
        copy_b128_lds(Bt + (size_t)(n0 + row) * K + kb1 + off,
                      &bT[ib ^ 1][row * 32 + off]);
      }
    }
#else
    uint4 ra[2], rb[4];                       // register-staged prefetch
    if (t + 1 < nk) {
#pragma unroll
      for (int it = 0; it < 2; ++it) {
        int c = tid + it * 256;
        int row = c >> 2, off = (c & 3) * 8;
        ra[it] = *(const uint4*)(A + (size_t)(m0 + row) * K + kb1 + off);
      }
#pragma unroll
      for (int it = 0; it < 4; ++it) {
        int c = tid + it * 256;
        int row = c >> 2, off = (c & 3) * 8;
        rb[it] = *(const uint4*)(Bt + (size_t)(n0 + row) * K + kb1 + off);
      }
    }
#endif

    // ---- compute on buffer ib: 16 WMMAs ----
    Frag aF[4];
#pragma unroll
    for (int i = 0; i < 4; ++i) {
      int row = wm * 64 + i * 16 + l15;
      aF[i].u4[0] = *(const uint4*)(&aT[ib][row * 32 + dsel]);
      aF[i].u4[1] = *(const uint4*)(&aT[ib][row * 32 + 16 + dsel]);
    }
#pragma unroll
    for (int j = 0; j < 4; ++j) {
      int col = wn * 64 + j * 16 + l15;
      Frag bF;
      bF.u4[0] = *(const uint4*)(&bT[ib][col * 32 + dsel]);
      bF.u4[1] = *(const uint4*)(&bT[ib][col * 32 + 16 + dsel]);
#pragma unroll
      for (int i = 0; i < 4; ++i)
        acc[i][j] = wmma_bf16(aF[i], bF, acc[i][j]);
    }

#if HAVE_ASYNC
    async_join();
#else
    if (t + 1 < nk) {
#pragma unroll
      for (int it = 0; it < 2; ++it) {
        int c = tid + it * 256;
        int row = c >> 2, off = (c & 3) * 8;
        *(uint4*)(&aT[ib ^ 1][row * 32 + off]) = ra[it];
      }
#pragma unroll
      for (int it = 0; it < 4; ++it) {
        int c = tid + it * 256;
        int row = c >> 2, off = (c & 3) * 8;
        *(uint4*)(&bT[ib ^ 1][row * 32 + off]) = rb[it];
      }
    }
#endif
    __syncthreads();
  }

  // ---- epilogue: bias + store ----
#pragma unroll
  for (int j = 0; j < 4; ++j) {
    int col  = n0 + wn * 64 + j * 16 + l15;
    float bv = bias[col];
#pragma unroll
    for (int i = 0; i < 4; ++i) {
#pragma unroll
      for (int e = 0; e < 8; ++e) {
        int row   = m0 + wm * 64 + i * 16 + e + hsel;
        float val = acc[i][j][e] + bv;
        if (OUT_F32) ((float*)Ov)[(size_t)row * N + col] = val;
        else ((unsigned short*)Ov)[(size_t)row * N + col] = f2bf(val);
      }
    }
  }
}

// ---------------------------------------------------------------------------
// Flash attention: block = (b, h, 128 query rows); 8 waves x 16-row strips.
// qkv bf16 [B*T, 3C]; y bf16 [B*T, C].
// ---------------------------------------------------------------------------
__global__ __launch_bounds__(256)
void attn_kernel(const unsigned short* __restrict__ qkv,
                 unsigned short* __restrict__ y) {
  __shared__ unsigned short kT[64 * 64];      // [key][d]
  __shared__ unsigned short vT[64 * 64];      // [d][key] (transposed)
  __shared__ unsigned short pS[8][16 * 64];   // per-wave P staging

  const int tid  = threadIdx.x;
  const int lane = tid & 31;
  const int wave = tid >> 5;
  const int l15  = lane & 15;
  const int dsel = (lane < 16) ? 0 : 8;
  const int rsel = (lane < 16) ? 0 : 16;
  const int hsel = (lane >> 4) << 3;

  const int bi = blockIdx.z;
  const int h  = blockIdx.y;
  const int q0 = blockIdx.x * 128;
  const int RS = 3 * CC;

  const int qrow = q0 + wave * 16 + l15;
  const size_t qBase = (size_t)(bi * TT + qrow) * RS + h * DD;
  Frag qF[2];
#pragma unroll
  for (int kk = 0; kk < 2; ++kk) {
    qF[kk].u4[0] = *(const uint4*)(qkv + qBase + kk * 32 + dsel);
    qF[kk].u4[1] = *(const uint4*)(qkv + qBase + kk * 32 + 16 + dsel);
  }

  float m[8], l[8];
  v8f o[4];
#pragma unroll
  for (int e = 0; e < 8; ++e) { m[e] = -3.0e38f; l[e] = 0.0f; }
#pragma unroll
  for (int n = 0; n < 4; ++n) o[n] = {};

  const float scale = 0.125f;
  const int kEnd = q0 + 128;

  for (int kb = 0; kb < kEnd; kb += 64) {
    // ---- stage K (async-capable) and V (manual transpose) ----
#pragma unroll
    for (int it = 0; it < 2; ++it) {
      int idx = tid + it * 256;
      int r = idx >> 3, c8 = (idx & 7) * 8;
      size_t g = (size_t)(bi * TT + kb + r) * RS + h * DD + c8;
      copy_b128_lds(qkv + g + CC, &kT[r * 64 + c8]);
      uint4 vd = *(const uint4*)(qkv + g + 2 * CC);
      const unsigned short* pv = (const unsigned short*)&vd;
#pragma unroll
      for (int j = 0; j < 8; ++j) vT[(c8 + j) * 64 + r] = pv[j];
    }
    // prefetch next K/V tile into cache
    if (kb + 64 < kEnd) {
      int r = tid >> 2, c16 = (tid & 3) * 16;
      size_t gp = (size_t)(bi * TT + kb + 64 + r) * RS + h * DD + c16;
      __builtin_prefetch((const void*)(qkv + gp + CC), 0, 0);
      __builtin_prefetch((const void*)(qkv + gp + 2 * CC), 0, 0);
    }
    async_join();
    __syncthreads();

    // ---- S = Q * K^T ----
    v8f s[4];
#pragma unroll
    for (int n = 0; n < 4; ++n) s[n] = {};
#pragma unroll
    for (int kk = 0; kk < 2; ++kk) {
#pragma unroll
      for (int n = 0; n < 4; ++n) {
        Frag kF;
        int kc = n * 16 + l15;
        kF.u4[0] = *(const uint4*)(&kT[kc * 64 + kk * 32 + dsel]);
        kF.u4[1] = *(const uint4*)(&kT[kc * 64 + kk * 32 + 16 + dsel]);
        s[n] = wmma_bf16(qF[kk], kF, s[n]);
      }
    }

    // ---- scale + causal mask ----
#pragma unroll
    for (int n = 0; n < 4; ++n) {
#pragma unroll
      for (int e = 0; e < 8; ++e) {
        float v  = s[n][e] * scale;
        int kcol = kb + n * 16 + l15;
        int qr   = q0 + wave * 16 + e + hsel;
        s[n][e]  = (kcol > qr) ? -3.0e38f : v;
      }
    }

    // ---- online softmax ----
    float mt[8];
#pragma unroll
    for (int e = 0; e < 8; ++e)
      mt[e] = fmaxf(fmaxf(s[0][e], s[1][e]), fmaxf(s[2][e], s[3][e]));
#pragma unroll
    for (int off = 1; off < 16; off <<= 1)
#pragma unroll
      for (int e = 0; e < 8; ++e)
        mt[e] = fmaxf(mt[e], __shfl_xor(mt[e], off, 32));

    float alpha[8], rs[8];
#pragma unroll
    for (int e = 0; e < 8; ++e) {
      float mn = fmaxf(m[e], mt[e]);
      alpha[e] = __expf(m[e] - mn);
      m[e]     = mn;
      rs[e]    = 0.0f;
    }
#pragma unroll
    for (int n = 0; n < 4; ++n)
#pragma unroll
      for (int e = 0; e < 8; ++e) {
        float p = __expf(s[n][e] - m[e]);
        s[n][e] = p;
        rs[e]  += p;
      }
#pragma unroll
    for (int off = 1; off < 16; off <<= 1)
#pragma unroll
      for (int e = 0; e < 8; ++e)
        rs[e] += __shfl_xor(rs[e], off, 32);
#pragma unroll
    for (int e = 0; e < 8; ++e) l[e] = l[e] * alpha[e] + rs[e];
#pragma unroll
    for (int n = 0; n < 4; ++n)
#pragma unroll
      for (int e = 0; e < 8; ++e) o[n][e] *= alpha[e];

    // ---- P: C-layout -> LDS -> A-fragments ----
#pragma unroll
    for (int n = 0; n < 4; ++n)
#pragma unroll
      for (int e = 0; e < 8; ++e)
        pS[wave][(e + hsel) * 64 + n * 16 + l15] = f2bf(s[n][e]);

    // ---- O += P * V ----
#pragma unroll
    for (int kk = 0; kk < 2; ++kk) {
      Frag pF;
      pF.u4[0] = *(const uint4*)(&pS[wave][l15 * 64 + kk * 32 + dsel]);
      pF.u4[1] = *(const uint4*)(&pS[wave][l15 * 64 + kk * 32 + 16 + dsel]);
#pragma unroll
      for (int n = 0; n < 4; ++n) {
        Frag vF;
        int dc = n * 16 + l15;
        vF.u4[0] = *(const uint4*)(&vT[dc * 64 + kk * 32 + dsel]);
        vF.u4[1] = *(const uint4*)(&vT[dc * 64 + kk * 32 + 16 + dsel]);
        o[n] = wmma_bf16(pF, vF, o[n]);
      }
    }
    __syncthreads();
  }

  // ---- normalize + store ----
#pragma unroll
  for (int n = 0; n < 4; ++n)
#pragma unroll
    for (int e = 0; e < 8; ++e) {
      int row = q0 + wave * 16 + e + hsel;
      int col = h * DD + n * 16 + l15;
      y[(size_t)(bi * TT + row) * CC + col] = f2bf(o[n][e] / l[e]);
    }
}

// ---------------------------------------------------------------------------
extern "C" void kernel_launch(void* const* d_in, const int* in_sizes, int n_in,
                              void* d_out, int out_size, void* d_ws, size_t ws_size,
                              hipStream_t stream) {
  (void)in_sizes; (void)n_in; (void)out_size; (void)ws_size;
  const float* x      = (const float*)d_in[0];
  const float* W_attn = (const float*)d_in[1];
  const float* b_attn = (const float*)d_in[2];
  const float* W_proj = (const float*)d_in[3];
  const float* b_proj = (const float*)d_in[4];
  float* out = (float*)d_out;

  const size_t MT = (size_t)BB * TT;          // 8192 rows
  unsigned short* qkv = (unsigned short*)d_ws;            // [MT, 3C]
  unsigned short* yb  = qkv + MT * 3 * CC;                // [MT, C]
  unsigned short* xb  = yb + MT * CC;                     // [MT, C]
  unsigned short* wta = xb + MT * CC;                     // [3C, C] (W_attn^T)
  unsigned short* wtp = wta + (size_t)3 * CC * CC;        // [C, C]  (W_proj^T)

  dim3 blk(256);
  // prep
  convert_f32_bf16<<<dim3((MT * CC / 4 + 255) / 256), blk, 0, stream>>>(
      x, xb, (int)(MT * CC / 4));
  transpose_f32_bf16<<<dim3(3 * CC / 64, CC / 64), blk, 0, stream>>>(
      W_attn, wta, CC, 3 * CC);
  transpose_f32_bf16<<<dim3(CC / 64, CC / 64), blk, 0, stream>>>(
      W_proj, wtp, CC, CC);
  // qkv = x @ W_attn + b_attn
  gemm_tn_bias<false><<<dim3(3 * CC / 256, MT / 128), blk, 0, stream>>>(
      xb, wta, b_attn, qkv, (int)MT, 3 * CC, CC);
  // attention
  attn_kernel<<<dim3(TT / 128, NHH, BB), blk, 0, stream>>>(qkv, yb);
  // out = y @ W_proj + b_proj
  gemm_tn_bias<true><<<dim3(CC / 256, MT / 128), blk, 0, stream>>>(
      yb, wtp, b_proj, out, (int)MT, CC, CC);
}
